// OA_1_14628658610727
// MI455X (gfx1250) — matrix operand order repaired and checked by reading, not compiled, not run on hardware
//
#include <hip/hip_runtime.h>
#include <hip/hip_bf16.h>
#include <stdint.h>

#define BB 16
#define CC 256
#define C2d 128
#define NNd 2048
#define BN_EPS 1e-5f

typedef __attribute__((ext_vector_type(16))) __bf16 v16bf;
typedef __attribute__((ext_vector_type(8)))  float  v8f;

union FragAB {
  v16bf v;
  unsigned short u[16];
  uint4 q[2];
};

__device__ __forceinline__ unsigned short f2bf(float f) {
  unsigned int xi = __builtin_bit_cast(unsigned int, f);
  xi += 0x7FFFu + ((xi >> 16) & 1u);           // round-to-nearest-even
  return (unsigned short)(xi >> 16);
}

__device__ __forceinline__ v8f wmma_bf16(const FragAB& a, const FragAB& b, v8f c) {
  return __builtin_amdgcn_wmma_f32_16x16x32_bf16(false, a.v, false, b.v,
                                                 (short)0, c, false, false);
}

// ---------------------------------------------------------------------------
// prep: zero column-sum buffer, convert ALL weight matrices -> bf16 once
// ---------------------------------------------------------------------------
__global__ void k_prep(const float* __restrict__ Wq, const float* __restrict__ Wk,
                       const float* __restrict__ Wv, const float* __restrict__ Wt,
                       unsigned short* __restrict__ WQ, unsigned short* __restrict__ WK,
                       unsigned short* __restrict__ WV, unsigned short* __restrict__ WT,
                       float* __restrict__ CS) {
  int i = blockIdx.x * blockDim.x + threadIdx.x;
  if (i < C2d * C2d) { WQ[i] = f2bf(Wq[i]); WK[i] = f2bf(Wk[i]); }
  if (i < CC * C2d)  { WV[i] = f2bf(Wv[i]); }
  if (i < CC * CC)   { WT[i] = f2bf(Wt[i]); }
  if (i < BB * NNd)  { CS[i] = 0.0f; }
}

// ---------------------------------------------------------------------------
// projections: x_q, x_k -> [B,N,C2] bf16 ; x_v -> [B,C,N] bf16 ; x2 -> [B,C,N] f32
// block = (b, 64-column n tile), 4 waves
// ---------------------------------------------------------------------------
#define P_NT 64
#define P_STR 136   // ushort row stride: 16B-aligned rows, conflict-friendly dword stride

__global__ __launch_bounds__(128) void k_project(
    const float* __restrict__ q, const float* __restrict__ x,
    const unsigned short* __restrict__ WQ, const unsigned short* __restrict__ WK,
    const unsigned short* __restrict__ WV, const float* __restrict__ bv,
    unsigned short* __restrict__ XQ, unsigned short* __restrict__ XK,
    unsigned short* __restrict__ XV, float* __restrict__ X2)
{
  __shared__ unsigned short sQ[P_NT * P_STR];
  __shared__ unsigned short sX[P_NT * P_STR];

  const int b   = blockIdx.x >> 5;            // 32 n-tiles per batch
  const int n0  = (blockIdx.x & 31) * P_NT;
  const int tid = threadIdx.x;

  // ---- stage 1: coalesced transpose-load q,x -> LDS bf16 [n][c] ----
  for (int it = tid; it < P_NT * (C2d / 2); it += 128) {
    int n  = it & (P_NT - 1);
    int cp = it >> 6;                          // c pair index 0..63
    int c  = cp * 2;
    size_t gq = ((size_t)b * C2d + c) * NNd + n0 + n;
    unsigned int pq = (unsigned int)f2bf(q[gq]) | ((unsigned int)f2bf(q[gq + NNd]) << 16);
    unsigned int px = (unsigned int)f2bf(x[gq]) | ((unsigned int)f2bf(x[gq + NNd]) << 16);
    *reinterpret_cast<unsigned int*>(&sQ[n * P_STR + c]) = pq;
    *reinterpret_cast<unsigned int*>(&sX[n * P_STR + c]) = px;
  }
  __syncthreads();

  const int lane = tid & 31;
  const int wv   = tid >> 5;
  const int ln   = lane & 15;
  const int g    = lane >> 4;
  const int noff = wv * 16;                    // 16-column n subtile per wave

  // ---- XQ / XK : A = bf16 weight rows (M=o), B = LDS tile (cols n) ----
  for (int sel = 0; sel < 2; ++sel) {
    const unsigned short* W = sel ? WK : WQ;
    const unsigned short* sT = sel ? sX : sQ;
    unsigned short* OUT = sel ? XK : XQ;
    for (int ot = 0; ot < 8; ++ot) {
      int o0 = ot * 16;
      v8f acc = {};
      #pragma unroll
      for (int k = 0; k < 4; ++k) {
        int k0 = k * 32;
        FragAB a, bf;
        const unsigned short* wr = W + (size_t)(o0 + ln) * C2d + k0;
        a.q[0] = *reinterpret_cast<const uint4*>(wr + 8 * g);
        a.q[1] = *reinterpret_cast<const uint4*>(wr + 16 + 8 * g);
        const unsigned short* br = sT + (noff + ln) * P_STR + k0;
        bf.q[0] = *reinterpret_cast<const uint4*>(br + 8 * g);
        bf.q[1] = *reinterpret_cast<const uint4*>(br + 16 + 8 * g);
        acc = wmma_bf16(a, bf, acc);
      }
      alignas(16) unsigned short pk[8];
      #pragma unroll
      for (int r = 0; r < 8; ++r) pk[r] = f2bf(acc[r]);
      unsigned short* dst = OUT + ((size_t)b * NNd + n0 + noff + ln) * C2d + o0 + 8 * g;
      *reinterpret_cast<uint4*>(dst) = *reinterpret_cast<const uint4*>(pk);
    }
  }

  // ---- XV / X2 : A = LDS tile rows (M=n), B = bf16 Wv rows (cols o) ----
  for (int sel = 0; sel < 2; ++sel) {
    const unsigned short* sT = sel ? sX : sQ;  // sel0: x_v from q ; sel1: x2 from x
    for (int ot = 0; ot < 16; ++ot) {
      int o0 = ot * 16;
      v8f acc = {};
      #pragma unroll
      for (int k = 0; k < 4; ++k) {
        int k0 = k * 32;
        FragAB a, bf;
        const unsigned short* ar = sT + (noff + ln) * P_STR + k0;
        a.q[0] = *reinterpret_cast<const uint4*>(ar + 8 * g);
        a.q[1] = *reinterpret_cast<const uint4*>(ar + 16 + 8 * g);
        const unsigned short* wr = WV + (size_t)(o0 + ln) * C2d + k0;
        bf.q[0] = *reinterpret_cast<const uint4*>(wr + 8 * g);
        bf.q[1] = *reinterpret_cast<const uint4*>(wr + 16 + 8 * g);
        acc = wmma_bf16(a, bf, acc);
      }
      float bias = bv[o0 + ln];
      if (sel == 0) {
        alignas(16) unsigned short pk[8];
        #pragma unroll
        for (int r = 0; r < 8; ++r) pk[r] = f2bf(acc[r] + bias);
        unsigned short* dst = XV + ((size_t)b * CC + o0 + ln) * NNd + n0 + noff + 8 * g;
        *reinterpret_cast<uint4*>(dst) = *reinterpret_cast<const uint4*>(pk);
      } else {
        alignas(16) float pf[8];
        #pragma unroll
        for (int r = 0; r < 8; ++r) pf[r] = acc[r] + bias;
        float* dst = X2 + ((size_t)b * CC + o0 + ln) * NNd + n0 + noff + 8 * g;
        *reinterpret_cast<float4*>(dst)     = *reinterpret_cast<const float4*>(pf);
        *reinterpret_cast<float4*>(dst + 4) = *reinterpret_cast<const float4*>(pf + 4);
      }
    }
  }
}

// ---------------------------------------------------------------------------
// energy + softmax: per (b, 16-row tile) compute [16 x 2048] energies into LDS,
// softmax over rows, write P^T [B][m][n] bf16 + atomic column sums
// ---------------------------------------------------------------------------
#define ESTR 2052   // padded f32 row stride (2052 % 64 == 4 -> conflict-free column reads)

__global__ __launch_bounds__(256) void k_energy_softmax(
    const unsigned short* __restrict__ XQ, const unsigned short* __restrict__ XK,
    unsigned short* __restrict__ PT, float* __restrict__ CS)
{
  __shared__ float sE[16 * ESTR];
  __shared__ float sRed[256];
  __shared__ float sMax[16];
  __shared__ float sSum[16];

  const int b   = blockIdx.x >> 7;
  const int n0  = (blockIdx.x & 127) * 16;
  const int tid = threadIdx.x;
  const int lane = tid & 31;
  const int wv  = tid >> 5;
  const int ln  = lane & 15;
  const int g   = lane >> 4;

  // preload A fragments (x_q rows n0..n0+15) once
  FragAB a[4];
  {
    const unsigned short* ar = XQ + ((size_t)b * NNd + n0 + ln) * C2d;
    #pragma unroll
    for (int k = 0; k < 4; ++k) {
      a[k].q[0] = *reinterpret_cast<const uint4*>(ar + k * 32 + 8 * g);
      a[k].q[1] = *reinterpret_cast<const uint4*>(ar + k * 32 + 16 + 8 * g);
    }
  }

  // each wave covers a 256-wide m stripe: 16 WMMA tiles of 16x16, K=128
  for (int j = 0; j < 16; ++j) {
    int m0 = wv * 256 + j * 16;
    const unsigned short* br = XK + ((size_t)b * NNd + m0 + ln) * C2d;
    v8f acc = {};
    #pragma unroll
    for (int k = 0; k < 4; ++k) {
      FragAB bf;
      bf.q[0] = *reinterpret_cast<const uint4*>(br + k * 32 + 8 * g);
      bf.q[1] = *reinterpret_cast<const uint4*>(br + k * 32 + 16 + 8 * g);
      acc = wmma_bf16(a[k], bf, acc);
    }
    #pragma unroll
    for (int r = 0; r < 8; ++r)
      sE[(r + 8 * g) * ESTR + m0 + ln] = acc[r];
  }
  __syncthreads();

  // row max (16 threads per row, 128 elements each)
  {
    int row = tid >> 4, seg = tid & 15;
    const float* er = sE + row * ESTR + seg * 128;
    float mx = -3.0e38f;
    for (int i = 0; i < 128; ++i) mx = fmaxf(mx, er[i]);
    sRed[tid] = mx;
  }
  __syncthreads();
  if (tid < 16) {
    float mx = sRed[tid * 16];
    for (int i = 1; i < 16; ++i) mx = fmaxf(mx, sRed[tid * 16 + i]);
    sMax[tid] = mx;
  }
  __syncthreads();
  // row exp-sum
  {
    int row = tid >> 4, seg = tid & 15;
    float mx = sMax[row];
    const float* er = sE + row * ESTR + seg * 128;
    float s = 0.f;
    for (int i = 0; i < 128; ++i) s += __expf(er[i] - mx);
    sRed[tid] = s;
  }
  __syncthreads();
  if (tid < 16) {
    float s = 0.f;
    for (int i = 0; i < 16; ++i) s += sRed[tid * 16 + i];
    sSum[tid] = s;
  }
  __syncthreads();

  float rmax[16], rinv[16];
  #pragma unroll
  for (int r = 0; r < 16; ++r) { rmax[r] = sMax[r]; rinv[r] = 1.0f / sSum[r]; }

  // write P^T[b][m][n0..n0+15] bf16 + partial column sums
  for (int m = tid; m < NNd; m += 256) {
    alignas(16) unsigned short pk[16];
    float csum = 0.f;
    #pragma unroll
    for (int r = 0; r < 16; ++r) {
      float p = __expf(sE[r * ESTR + m] - rmax[r]) * rinv[r];
      csum += p;
      pk[r] = f2bf(p);
    }
    unsigned short* dst = PT + ((size_t)b * NNd + m) * NNd + n0;
    *reinterpret_cast<uint4*>(dst)     = *reinterpret_cast<const uint4*>(pk);
    *reinterpret_cast<uint4*>(dst + 8) = *reinterpret_cast<const uint4*>(pk + 8);
    atomicAdd(&CS[b * NNd + m], csum);
  }
}

// ---------------------------------------------------------------------------
// x_r = XV * P (K = N = 2048), apply column L1 renorm, diff = x2 - x_r -> bf16 [B,N,C]
// block tile 32c x 64m, 8 waves
// ---------------------------------------------------------------------------
__global__ __launch_bounds__(256) void k_xr_diff(
    const unsigned short* __restrict__ XV, const unsigned short* __restrict__ PT,
    const float* __restrict__ CS, const float* __restrict__ X2,
    unsigned short* __restrict__ DIFF)
{
  const int b    = blockIdx.z;
  const int tid  = threadIdx.x;
  const int lane = tid & 31;
  const int wv   = tid >> 5;
  const int ln   = lane & 15, g = lane >> 4;
  const int c0   = blockIdx.x * 32 + (wv & 1) * 16;
  const int m0   = blockIdx.y * 64 + (wv >> 1) * 16;

  const unsigned short* ar = XV + ((size_t)b * CC  + c0 + ln) * NNd;
  const unsigned short* br = PT + ((size_t)b * NNd + m0 + ln) * NNd;

  v8f acc = {};
  #pragma unroll 4
  for (int kk = 0; kk < 64; ++kk) {
    FragAB a, bf;
    int kb = kk * 32;
    if (kk + 8 < 64) {                     // stream-ahead prefetch (global_prefetch_b8)
      __builtin_prefetch(ar + (kk + 8) * 32, 0, 0);
      __builtin_prefetch(br + (kk + 8) * 32, 0, 0);
    }
    a.q[0]  = *reinterpret_cast<const uint4*>(ar + kb + 8 * g);
    a.q[1]  = *reinterpret_cast<const uint4*>(ar + kb + 16 + 8 * g);
    bf.q[0] = *reinterpret_cast<const uint4*>(br + kb + 8 * g);
    bf.q[1] = *reinterpret_cast<const uint4*>(br + kb + 16 + 8 * g);
    acc = wmma_bf16(a, bf, acc);
  }

  int m = m0 + ln;
  float rinv = 1.0f / (1e-9f + CS[b * NNd + m]);
  alignas(16) unsigned short pk[8];
  #pragma unroll
  for (int r = 0; r < 8; ++r) {
    int c = c0 + 8 * g + r;
    float xr  = acc[r] * rinv;
    float x2v = X2[((size_t)b * CC + c) * NNd + m];
    pk[r] = f2bf(x2v - xr);
  }
  unsigned short* dst = DIFF + ((size_t)b * NNd + m) * CC + c0 + 8 * g;
  *reinterpret_cast<uint4*>(dst) = *reinterpret_cast<const uint4*>(pk);
}

// ---------------------------------------------------------------------------
// t = Wt*(diff) + bt -> BN -> ReLU -> + x2 ; out [B,C,N] f32
// block tile 32o x 64n, 8 waves, K = C = 256
// ---------------------------------------------------------------------------
__global__ __launch_bounds__(256) void k_final(
    const unsigned short* __restrict__ WT, const unsigned short* __restrict__ DIFF,
    const float* __restrict__ X2,
    const float* __restrict__ bt, const float* __restrict__ gamma,
    const float* __restrict__ beta, const float* __restrict__ mean,
    const float* __restrict__ var, float* __restrict__ out)
{
  const int b    = blockIdx.z;
  const int tid  = threadIdx.x;
  const int lane = tid & 31;
  const int wv   = tid >> 5;
  const int ln   = lane & 15, g = lane >> 4;
  const int o0   = blockIdx.x * 32 + (wv & 1) * 16;
  const int n0   = blockIdx.y * 64 + (wv >> 1) * 16;

  const unsigned short* ar = WT + (size_t)(o0 + ln) * CC;
  const unsigned short* br = DIFF + ((size_t)b * NNd + n0 + ln) * CC;

  v8f acc = {};
  #pragma unroll
  for (int k = 0; k < 8; ++k) {
    FragAB a, bf;
    int kb = k * 32;
    a.q[0]  = *reinterpret_cast<const uint4*>(ar + kb + 8 * g);
    a.q[1]  = *reinterpret_cast<const uint4*>(ar + kb + 16 + 8 * g);
    bf.q[0] = *reinterpret_cast<const uint4*>(br + kb + 8 * g);
    bf.q[1] = *reinterpret_cast<const uint4*>(br + kb + 16 + 8 * g);
    acc = wmma_bf16(a, bf, acc);
  }

  int n = n0 + ln;
  #pragma unroll
  for (int r = 0; r < 8; ++r) {
    int o = o0 + 8 * g + r;
    float t   = acc[r] + bt[o];
    float inv = gamma[o] * rsqrtf(var[o] + BN_EPS);
    float bn  = t * inv + (beta[o] - mean[o] * inv);
    float x2v = X2[((size_t)b * CC + o) * NNd + n];
    out[((size_t)b * CC + o) * NNd + n] = fmaxf(bn, 0.0f) + x2v;
  }
}

// ---------------------------------------------------------------------------
extern "C" void kernel_launch(void* const* d_in, const int* in_sizes, int n_in,
                              void* d_out, int out_size, void* d_ws, size_t ws_size,
                              hipStream_t stream) {
  (void)in_sizes; (void)n_in; (void)out_size; (void)ws_size;
  const float* q     = (const float*)d_in[0];
  const float* x     = (const float*)d_in[1];
  const float* Wq    = (const float*)d_in[2];
  const float* Wk    = (const float*)d_in[3];
  const float* Wv    = (const float*)d_in[4];
  const float* bv    = (const float*)d_in[5];
  const float* Wt    = (const float*)d_in[6];
  const float* bt    = (const float*)d_in[7];
  const float* gamma = (const float*)d_in[8];
  const float* beta  = (const float*)d_in[9];
  const float* rmean = (const float*)d_in[10];
  const float* rvar  = (const float*)d_in[11];

  char* ws = (char*)d_ws;
  size_t off = 0;
  auto alloc = [&](size_t bytes) -> void* {
    void* p = ws + off;
    off = (off + bytes + 255) & ~(size_t)255;
    return p;
  };
  unsigned short* XQ   = (unsigned short*)alloc((size_t)BB * NNd * C2d * 2); //  8 MB
  unsigned short* XK   = (unsigned short*)alloc((size_t)BB * NNd * C2d * 2); //  8 MB
  unsigned short* XV   = (unsigned short*)alloc((size_t)BB * CC * NNd * 2);  // 16 MB
  float*          X2   = (float*)alloc((size_t)BB * CC * NNd * 4);           // 32 MB
  unsigned short* DIF  = (unsigned short*)alloc((size_t)BB * NNd * CC * 2);  // 16 MB
  unsigned short* WQ16 = (unsigned short*)alloc((size_t)C2d * C2d * 2);      // 32 KB
  unsigned short* WK16 = (unsigned short*)alloc((size_t)C2d * C2d * 2);      // 32 KB
  unsigned short* WV16 = (unsigned short*)alloc((size_t)CC * C2d * 2);       // 64 KB
  unsigned short* WT16 = (unsigned short*)alloc((size_t)CC * CC * 2);        // 128 KB
  float*          CS   = (float*)alloc((size_t)BB * NNd * 4);                // 128 KB
  unsigned short* PT   = (unsigned short*)alloc((size_t)BB * NNd * NNd * 2); // 128 MB

  k_prep<<<(CC * CC + 255) / 256, 256, 0, stream>>>(Wq, Wk, Wv, Wt,
                                                    WQ16, WK16, WV16, WT16, CS);
  k_project<<<BB * (NNd / P_NT), 128, 0, stream>>>(q, x, WQ16, WK16, WV16, bv,
                                                   XQ, XK, XV, X2);
  k_energy_softmax<<<BB * (NNd / 16), 256, 0, stream>>>(XQ, XK, PT, CS);
  dim3 gR(CC / 32, NNd / 64, BB);
  k_xr_diff<<<gR, 256, 0, stream>>>(XV, PT, CS, X2, DIF);
  dim3 gT(CC / 32, NNd / 64, BB);
  k_final<<<gT, 256, 0, stream>>>(WT16, DIF, X2, bt, gamma, beta, rmean, rvar,
                                  (float*)d_out);
}
